// Model_11647951307477
// MI455X (gfx1250) — compile-verified, matrix operand order checked by reading, not run on hardware
//
#include <hip/hip_runtime.h>
#include <math.h>

// ---------------------------------------------------------------------------
// Types
// ---------------------------------------------------------------------------
typedef _Float16 half_t;
typedef __attribute__((ext_vector_type(8)))  _Float16 v8h;
typedef __attribute__((ext_vector_type(16))) _Float16 v16h;
typedef __attribute__((ext_vector_type(8)))  float    v8f;

#define BN_EPS_F 1e-5f
#define PD_EPS_F 1e-6f
#define MARGIN_F 0.1f

// ---------------------------------------------------------------------------
// BN fold: Wt[co][k] = w[k][co] * scale(co)   (fp32 HWIO -> fp16 [Cout][K])
//          bias[co]  = b[co] - m[co]*scale(co)
// ---------------------------------------------------------------------------
__global__ void fold_bn_kernel(const float* __restrict__ w,
                               const float* __restrict__ g,
                               const float* __restrict__ bb,
                               const float* __restrict__ m,
                               const float* __restrict__ v,
                               half_t* __restrict__ Wt,
                               float* __restrict__ bias,
                               int K, int Cout) {
  int idx = blockIdx.x * blockDim.x + threadIdx.x;
  if (idx >= K * Cout) return;
  int co = idx / K;
  int k  = idx - co * K;
  float s = g[co] * rsqrtf(v[co] + BN_EPS_F);
  Wt[idx] = (half_t)(w[(size_t)k * Cout + co] * s);
  if (k == 0) bias[co] = bb[co] - m[co] * s;
}

// ---------------------------------------------------------------------------
// Implicit-GEMM conv via V_WMMA_F32_16X16X32_F16, software-pipelined with
// double-buffered LDS (one barrier per K-step; global fetch of chunk k+1
// overlaps the 4 WMMAs of chunk k).
// X: NHWC fp16 [B,IH,IW,Cin]; Wt: fp16 [Cout][KH*KW*Cin] (k = (kh,kw,ci));
// Y/Res: NHWC fp16 [M=B*OH*OW][Cout]. Epilogue: +bias, +Res, ReLU.
// Block = 256 threads (8 wave32s): tile M=128 x N=64, K-step 32.
// Wave grid 4(M) x 2(N); each wave owns 32x32 = 2x2 accumulators.
// Requires Cin % 32 == 0 and Cout % 64 == 0 (true for all bottleneck convs).
// ---------------------------------------------------------------------------
__global__ __launch_bounds__(256)
void conv_wmma_kernel(const half_t* __restrict__ X,
                      const half_t* __restrict__ Wt,
                      const float*  __restrict__ bias,
                      const half_t* __restrict__ Res,
                      half_t* __restrict__ Y,
                      int M_, int IH, int IW, int Cin,
                      int OH, int OW, int Cout,
                      int KW, int stride, int pad, int Ktot, int relu) {
  __shared__ __align__(16) half_t As[2][128][40];  // 2 x (128 rows x 32 K, +8 pad)
  __shared__ __align__(16) half_t Bs[2][64][40];   // 2 x (64 N x 32 K, +8 pad)

  const int tid  = threadIdx.x;
  const int lane = tid & 31;
  const int wv   = tid >> 5;       // wave id 0..7
  const int wm   = wv & 3;         // M strip (32 rows each)
  const int wn   = wv >> 2;        // N strip (32 cols each)
  const int m0   = blockIdx.x * 128;
  const int n0   = blockIdx.y * 64;

  // --- A cooperative-load state: each thread owns rows ar and ar+64, one
  //     8-half K chunk per row per K-step ---
  const int ar = tid >> 2;               // 0..63  (second row = ar+64)
  const int ac = (tid & 3) * 8;          // K offset within 32-chunk

  int ohs[2], ows[2], bidx[2];
  bool arow_ok[2];
#pragma unroll
  for (int r = 0; r < 2; ++r) {
    const int am = m0 + ar + r * 64;
    arow_ok[r] = (am < M_);
    int mm = arow_ok[r] ? am : 0;
    int ow = mm % OW;  int t1 = mm / OW;
    int oh = t1 % OH;  bidx[r] = t1 / OH;
    ohs[r] = oh * stride - pad;
    ows[r] = ow * stride - pad;
  }

  // k-decode state for the NEXT chunk to be fetched (incremental, no divides;
  // Cin % 32 == 0 so a 32-chunk never straddles a (kh,kw) boundary)
  int ci = ac, kh = 0, kw = 0;
  const half_t* wp = Wt + (size_t)(n0 + ar) * Ktot + ac;

  auto fetch = [&](v8h& a0v, v8h& a1v, v8h& bv) {
#pragma unroll
    for (int r = 0; r < 2; ++r) {
      v8h z = (v8h){0, 0, 0, 0, 0, 0, 0, 0};
      const int ih = ohs[r] + kh, iw = ows[r] + kw;
      if (arow_ok[r] && ih >= 0 && ih < IH && iw >= 0 && iw < IW) {
        const half_t* xp = X + (((size_t)bidx[r] * IH + ih) * IW + iw) * Cin + ci;
        z = *(const v8h*)xp;             // global_load_b128
      }
      if (r == 0) a0v = z; else a1v = z;
    }
    bv = *(const v8h*)wp;                // global_load_b128
  };
  auto advance = [&]() {
    ci += 32;
    if (ci >= Cin) { ci -= Cin; kw++; if (kw == KW) { kw = 0; kh++; } }
    wp += 32;
  };

  v8f acc[2][2];
#pragma unroll
  for (int i = 0; i < 2; ++i)
#pragma unroll
    for (int j = 0; j < 2; ++j)
      acc[i][j] = (v8f){0.f, 0.f, 0.f, 0.f, 0.f, 0.f, 0.f, 0.f};

  const int steps = Ktot >> 5;

  // ---- prologue: stage chunk 0 into buffer 0 ----
  {
    v8h pa0, pa1, pb;
    fetch(pa0, pa1, pb);
    *(v8h*)&As[0][ar][ac]      = pa0;    // ds_store_b128
    *(v8h*)&As[0][ar + 64][ac] = pa1;
    *(v8h*)&Bs[0][ar][ac]      = pb;
    advance();
  }
  __syncthreads();

  // ---- pipelined main loop: one barrier per K-step ----
  for (int s = 0; s < steps; ++s) {
    const int  cb   = s & 1;
    const bool more = (s + 1 < steps);
    v8h na0, na1, nb;
    if (more) {
      fetch(na0, na1, nb);               // overlaps WMMAs below
      __builtin_prefetch((const void*)(wp + 128), 0, 1);  // global_prefetch
    }

    // Fragment loads per CDNA5 16-bit A/B layout:
    //   lane<16:  elems 0..7 = K 0..7,  8..15 = K 16..23
    //   lane>=16: elems 0..7 = K 8..15, 8..15 = K 24..31
    const int kb = (lane >> 4) << 3;
    v16h af[2], bf[2];
#pragma unroll
    for (int i = 0; i < 2; ++i) {
      const int mr = wm * 32 + i * 16 + (lane & 15);
      v8h a0 = *(const v8h*)&As[cb][mr][kb];
      v8h a1 = *(const v8h*)&As[cb][mr][kb + 16];
      af[i] = __builtin_shufflevector(a0, a1, 0,1,2,3,4,5,6,7,8,9,10,11,12,13,14,15);
    }
#pragma unroll
    for (int j = 0; j < 2; ++j) {
      const int nr = wn * 32 + j * 16 + (lane & 15);
      v8h b0 = *(const v8h*)&Bs[cb][nr][kb];
      v8h b1 = *(const v8h*)&Bs[cb][nr][kb + 16];
      bf[j] = __builtin_shufflevector(b0, b1, 0,1,2,3,4,5,6,7,8,9,10,11,12,13,14,15);
    }

    if (more) {                          // deposit chunk s+1 into other buffer
      *(v8h*)&As[cb ^ 1][ar][ac]      = na0;
      *(v8h*)&As[cb ^ 1][ar + 64][ac] = na1;
      *(v8h*)&Bs[cb ^ 1][ar][ac]      = nb;
      advance();
    }

#pragma unroll
    for (int i = 0; i < 2; ++i)
#pragma unroll
      for (int j = 0; j < 2; ++j)
        acc[i][j] = __builtin_amdgcn_wmma_f32_16x16x32_f16(
            false, af[i], false, bf[j], (short)0, acc[i][j], false, false);

    __syncthreads();
  }

  // Epilogue: C/D layout -> lane n = lane%16; VGPR e holds M = e + 8*(lane/16)
#pragma unroll
  for (int i = 0; i < 2; ++i) {
    const int mb = m0 + wm * 32 + i * 16 + ((lane >> 4) << 3);
#pragma unroll
    for (int j = 0; j < 2; ++j) {
      const int ng = n0 + wn * 32 + j * 16 + (lane & 15);
      const float bz = bias[ng];
#pragma unroll
      for (int e = 0; e < 8; ++e) {
        const int mg = mb + e;
        if (mg < M_) {
          float val = acc[i][j][e] + bz;
          if (Res) val += (float)Res[(size_t)mg * Cout + ng];
          if (relu) val = fmaxf(val, 0.0f);
          Y[(size_t)mg * Cout + ng] = (half_t)val;
        }
      }
    }
  }
}

// ---------------------------------------------------------------------------
// Stem: 7x7 s2 p3 conv (Cin=3) direct, fp32 image NCHW -> fp16 NHWC, BN+ReLU
// ---------------------------------------------------------------------------
__global__ void stem_kernel(const float* __restrict__ X,
                            const float* __restrict__ Wf,
                            const float* __restrict__ g,
                            const float* __restrict__ bb,
                            const float* __restrict__ m,
                            const float* __restrict__ v,
                            half_t* __restrict__ Y) {
  int idx = blockIdx.x * blockDim.x + threadIdx.x;
  const int total = 32 * 112 * 112 * 64;
  if (idx >= total) return;
  int c = idx & 63;  int t = idx >> 6;
  int ow = t % 112;  t /= 112;
  int oh = t % 112;  int b = t / 112;
  float s    = g[c] * rsqrtf(v[c] + BN_EPS_F);
  float bias = bb[c] - m[c] * s;
  float acc = 0.0f;
  for (int kh = 0; kh < 7; ++kh) {
    int ih = oh * 2 - 3 + kh;
    if (ih < 0 || ih >= 224) continue;
    for (int kw = 0; kw < 7; ++kw) {
      int iw = ow * 2 - 3 + kw;
      if (iw < 0 || iw >= 224) continue;
#pragma unroll
      for (int ci = 0; ci < 3; ++ci)
        acc += X[(((size_t)b * 3 + ci) * 224 + ih) * 224 + iw]
             * Wf[((kh * 7 + kw) * 3 + ci) * 64 + c];
    }
  }
  Y[idx] = (half_t)fmaxf(acc * s + bias, 0.0f);
}

// 3x3 s2 p1 maxpool: 112 -> 56, fp16 NHWC
__global__ void maxpool_kernel(const half_t* __restrict__ X, half_t* __restrict__ Y) {
  int idx = blockIdx.x * blockDim.x + threadIdx.x;
  const int total = 32 * 56 * 56 * 64;
  if (idx >= total) return;
  int c = idx & 63;  int t = idx >> 6;
  int ow = t % 56;   t /= 56;
  int oh = t % 56;   int b = t / 56;
  float mx = -INFINITY;
  for (int kh = 0; kh < 3; ++kh) {
    int ih = oh * 2 - 1 + kh;
    if (ih < 0 || ih >= 112) continue;
    for (int kw = 0; kw < 3; ++kw) {
      int iw = ow * 2 - 1 + kw;
      if (iw < 0 || iw >= 112) continue;
      mx = fmaxf(mx, (float)X[(((size_t)b * 112 + ih) * 112 + iw) * 64 + c]);
    }
  }
  Y[idx] = (half_t)mx;
}

// Global average pool: [32,7,7,2048] fp16 -> [32,2048] fp32
__global__ void avgpool_kernel(const half_t* __restrict__ X, float* __restrict__ P) {
  int idx = blockIdx.x * blockDim.x + threadIdx.x;
  if (idx >= 32 * 2048) return;
  int c = idx & 2047;  int b = idx >> 11;
  const half_t* xp = X + (size_t)b * 49 * 2048 + c;
  float s = 0.0f;
  for (int i = 0; i < 49; ++i) s += (float)xp[(size_t)i * 2048];
  P[idx] = s * (1.0f / 49.0f);
}

// FC: emb[b][j] = sum_k pooled[b][k] * W[k][j] + bias[j]   (fp32)
__global__ void fc_kernel(const float* __restrict__ P, const float* __restrict__ W,
                          const float* __restrict__ bvec, float* __restrict__ E) {
  int idx = blockIdx.x * blockDim.x + threadIdx.x;
  if (idx >= 32 * 300) return;
  int j = idx % 300;  int b = idx / 300;
  const float* x = P + (size_t)b * 2048;
  float acc = bvec[j];
  for (int k = 0; k < 2048; ++k) acc += x[k] * W[(size_t)k * 300 + j];
  E[idx] = acc;
}

__global__ void init_count_kernel(int* c) { *c = 0; }

// Pairwise distances + margin compare + count
__global__ void dist_count_kernel(const float* __restrict__ emb,
                                  const float* __restrict__ tp,
                                  const float* __restrict__ tn,
                                  int* __restrict__ count) {
  const int b = blockIdx.x;     // 0..31
  const int t = threadIdx.x;    // 128 threads
  __shared__ float s1[128];
  __shared__ float s2[128];
  float a1 = 0.0f, a2 = 0.0f;
  for (int j = t; j < 300; j += 128) {
    float e  = emb[b * 300 + j];
    float d1 = e - tp[b * 300 + j] + PD_EPS_F;
    float d2 = e - tn[b * 300 + j] + PD_EPS_F;
    a1 += d1 * d1;  a2 += d2 * d2;
  }
  s1[t] = a1;  s2[t] = a2;
  __syncthreads();
  for (int o = 64; o > 0; o >>= 1) {
    if (t < o) { s1[t] += s1[t + o]; s2[t] += s2[t + o]; }
    __syncthreads();
  }
  if (t == 0) {
    float dtp = sqrtf(s1[0]), dtn = sqrtf(s2[0]);
    if (dtn - dtp > MARGIN_F) atomicAdd(count, 1);
  }
}

// ---------------------------------------------------------------------------
// Host launcher
// ---------------------------------------------------------------------------
extern "C" void kernel_launch(void* const* d_in, const int* in_sizes, int n_in,
                              void* d_out, int out_size, void* d_ws, size_t ws_size,
                              hipStream_t stream) {
  (void)in_sizes; (void)n_in; (void)out_size; (void)ws_size;

  const float* image  = (const float*)d_in[0];   // [32,3,224,224]
  const float* tags_p = (const float*)d_in[1];   // [32,300]
  const float* tags_n = (const float*)d_in[2];   // [32,300]

  // params leaves: jax tree_flatten order (dict keys sorted, lists in order):
  // fc_b, fc_w, layers[ stage ][ block ]{ bn1(b,g,m,v), bn2, bn3, [bns],
  //   w1, w2, w3, [ws] }, stem_bn(b,g,m,v), stem_w
  int p = 3;
  auto nxt = [&]() { return (const float*)d_in[p++]; };

  struct BNp { const float *b, *g, *m, *v; };
  struct Blk { BNp bn1, bn2, bn3, bns; const float *w1, *w2, *w3, *ws; };

  const float* fc_b = nxt();
  const float* fc_w = nxt();

  Blk blocks[16];
  const int nb_stage[4] = {3, 4, 6, 3};
  int bi = 0;
  for (int s = 0; s < 4; ++s)
    for (int i = 0; i < nb_stage[s]; ++i) {
      Blk& B = blocks[bi++];
      auto getbn = [&]() { BNp r; r.b = nxt(); r.g = nxt(); r.m = nxt(); r.v = nxt(); return r; };
      B.bn1 = getbn();  B.bn2 = getbn();  B.bn3 = getbn();
      if (i == 0) B.bns = getbn();
      B.w1 = nxt();  B.w2 = nxt();  B.w3 = nxt();
      B.ws = (i == 0) ? nxt() : nullptr;
    }
  BNp stem_bn;
  stem_bn.b = nxt(); stem_bn.g = nxt(); stem_bn.m = nxt(); stem_bn.v = nxt();
  const float* stem_w = nxt();

  // ---- workspace bump allocator (deterministic) ----
  size_t off = 0;
  auto walloc = [&](size_t bytes) -> void* {
    off = (off + 255) & ~(size_t)255;
    void* ptr = (char*)d_ws + off;
    off += bytes;
    return ptr;
  };
  half_t* slabA = (half_t*)walloc((size_t)32 * 112 * 112 * 64 * 2);  // 51.4 MB
  half_t* slabD = (half_t*)walloc((size_t)32 * 112 * 112 * 64 * 2);  // 51.4 MB
  half_t* slabB = (half_t*)walloc((size_t)32 * 56 * 56 * 128 * 2);   // 25.7 MB
  half_t* slabC = (half_t*)walloc((size_t)32 * 56 * 56 * 64 * 2);    // 12.9 MB
  float*  pooled = (float*)walloc((size_t)32 * 2048 * 4);

  // fold-BN + implicit-GEMM conv launcher
  auto conv = [&](const half_t* X, const float* w, BNp bn,
                  half_t* Y, const half_t* Res,
                  int IH, int IW, int Cin, int OH, int OW, int Cout,
                  int KH, int KW, int stride, int pad, int relu) {
    const int K = KH * KW * Cin;
    half_t* Wt   = (half_t*)walloc((size_t)K * Cout * 2);
    float*  bias = (float*)walloc((size_t)Cout * 4);
    const int tot = K * Cout;
    fold_bn_kernel<<<dim3((tot + 255) / 256), dim3(256), 0, stream>>>(
        w, bn.g, bn.b, bn.m, bn.v, Wt, bias, K, Cout);
    const int M = 32 * OH * OW;
    dim3 grid((M + 127) / 128, Cout / 64);
    conv_wmma_kernel<<<grid, dim3(256), 0, stream>>>(
        X, Wt, bias, Res, Y, M, IH, IW, Cin, OH, OW, Cout,
        KW, stride, pad, K, relu);
  };

  // ---- stem + maxpool ----
  {
    const int tot = 32 * 112 * 112 * 64;
    stem_kernel<<<dim3((tot + 255) / 256), dim3(256), 0, stream>>>(
        image, stem_w, stem_bn.g, stem_bn.b, stem_bn.m, stem_bn.v, slabD);
    const int tp = 32 * 56 * 56 * 64;
    maxpool_kernel<<<dim3((tp + 255) / 256), dim3(256), 0, stream>>>(slabD, slabA);
  }

  // ---- 16 bottleneck blocks ----
  half_t* cur = slabA;
  half_t* alt = slabD;
  int H = 56, W = 56, Cin = 64;
  const int widths[4]  = {64, 128, 256, 512};
  const int strides[4] = {1, 2, 2, 2};
  bi = 0;
  for (int s = 0; s < 4; ++s) {
    const int w_ = widths[s];
    for (int i = 0; i < nb_stage[s]; ++i) {
      Blk& B = blocks[bi++];
      const int st = (i == 0) ? strides[s] : 1;
      const int OH = (H - 1) / st + 1;
      const int OW = (W - 1) / st + 1;
      const int Cout = 4 * w_;
      half_t* out = alt;
      const half_t* res;
      if (i == 0) {  // projection shortcut into 'out'; conv3 adds it back
        conv(cur, B.ws, B.bns, out, nullptr, H, W, Cin, OH, OW, Cout, 1, 1, st, 0, 0);
        res = out;
      } else {
        res = cur;   // identity shortcut
      }
      conv(cur,   B.w1, B.bn1, slabB, nullptr, H, W, Cin, H,  W,  w_,   1, 1, 1,  0, 1);
      conv(slabB, B.w2, B.bn2, slabC, nullptr, H, W, w_,  OH, OW, w_,   3, 3, st, 1, 1);
      conv(slabC, B.w3, B.bn3, out,   res,     OH, OW, w_, OH, OW, Cout, 1, 1, 1, 0, 1);
      half_t* tmp = cur; cur = out; alt = tmp;
      Cin = Cout;  H = OH;  W = OW;
    }
  }

  // ---- avgpool -> FC -> distances/count ----
  avgpool_kernel<<<dim3((32 * 2048 + 255) / 256), dim3(256), 0, stream>>>(cur, pooled);
  float* emb = (float*)d_out;                       // [32,300] fp32
  fc_kernel<<<dim3((32 * 300 + 255) / 256), dim3(256), 0, stream>>>(pooled, fc_w, fc_b, emb);
  int* count = (int*)(emb + 32 * 300);              // int32 "correct", shape (1,)
  init_count_kernel<<<dim3(1), dim3(1), 0, stream>>>(count);
  dist_count_kernel<<<dim3(32), dim3(128), 0, stream>>>(emb, tags_p, tags_n, count);
}
